// MixtralSparseMoeBlock_31937376813773
// MI455X (gfx1250) — matrix operand order repaired and checked by reading, not compiled, not run on hardware
//
#include <hip/hip_runtime.h>
#include <hip/hip_bf16.h>

typedef __attribute__((ext_vector_type(16))) __bf16 v16bf;
typedef __attribute__((ext_vector_type(8)))  float  v8f;
typedef __attribute__((ext_vector_type(4)))  int    v4i;

#define Dc 1024
#define Hc 2048
#define Ec 8
#define TM 128
#define TN 128
#define TK 32
#define BN_EPS 1e-5f

#if __has_builtin(__builtin_amdgcn_global_load_async_to_lds_b128)
#define HAS_ASYNC 1
#define NBUF 2
#else
#define HAS_ASYNC 0
#define NBUF 1
#endif

__device__ __forceinline__ unsigned short f2bf(float f) {
  unsigned u = __float_as_uint(f);
  u += 0x7FFFu + ((u >> 16) & 1u);          // round-to-nearest-even
  return (unsigned short)(u >> 16);
}

struct Frag { union { v16bf v; uint4 q[2]; }; };

__device__ __forceinline__ void async_b128(const unsigned short* g, unsigned short* l) {
#if HAS_ASYNC
  __builtin_amdgcn_global_load_async_to_lds_b128(
      (__attribute__((address_space(1))) v4i*)(unsigned short*)g,
      (__attribute__((address_space(3))) v4i*)l, 0, 0);
#endif
}

__device__ __forceinline__ void wait_async0() {
#if HAS_ASYNC
#if __has_builtin(__builtin_amdgcn_s_wait_asynccnt)
  __builtin_amdgcn_s_wait_asynccnt(0);
#else
  asm volatile("s_wait_asynccnt 0x0" ::: "memory");
#endif
#endif
}

// one 64x64 wave tile: 4x4 WMMA accumulate from LDS tiles [m][k] / [n][k]
__device__ __forceinline__ void wave_mma(const unsigned short* Al,
                                         const unsigned short* Bl,
                                         int wr, int wc, int l16, int hi,
                                         v8f acc[16]) {
  Frag Bf[4];
#pragma unroll
  for (int ni = 0; ni < 4; ++ni) {
    int n = wc * 64 + ni * 16 + l16;
    int kb = hi ? 16 : 0;
    Bf[ni].q[0] = *(const uint4*)&Bl[n * TK + kb];
    Bf[ni].q[1] = *(const uint4*)&Bl[n * TK + kb + 8];
  }
#pragma unroll
  for (int mi = 0; mi < 4; ++mi) {
    Frag Af;
    int m = wr * 64 + mi * 16 + l16;
    int kh = hi ? 16 : 0;
    Af.q[0] = *(const uint4*)&Al[m * TK + kh];
    Af.q[1] = *(const uint4*)&Al[m * TK + kh + 8];
#pragma unroll
    for (int ni = 0; ni < 4; ++ni)
      acc[mi * 4 + ni] = __builtin_amdgcn_wmma_f32_16x16x32_bf16(
          false, Af.v, false, Bf[ni].v, (short)0, acc[mi * 4 + ni], false, false);
  }
}

// shared GEMM mainloop: A row ga[K], B row gb[K] per thread (thread r stages row r)
template <int K>
__device__ __forceinline__ void gemm_mainloop(
    const unsigned short* __restrict__ ga, const unsigned short* __restrict__ gb,
    unsigned short (*Al)[TM * TK], unsigned short (*Bl)[TN * TK],
    int r, int wr, int wc, int l16, int hi, v8f acc[16]) {
#if HAS_ASYNC
  // double-buffered async pipeline: copy tile k+1 while WMMAs consume tile k
  async_b128(ga,     &Al[0][r * TK]);
  async_b128(ga + 8, &Al[0][r * TK + 8]);
  async_b128(ga + 16, &Al[0][r * TK + 16]);
  async_b128(ga + 24, &Al[0][r * TK + 24]);
  async_b128(gb,     &Bl[0][r * TK]);
  async_b128(gb + 8, &Bl[0][r * TK + 8]);
  async_b128(gb + 16, &Bl[0][r * TK + 16]);
  async_b128(gb + 24, &Bl[0][r * TK + 24]);
  int buf = 0;
  for (int k0 = 0; k0 < K; k0 += TK) {
    wait_async0();
    __syncthreads();                       // tile `buf` ready for all waves
    if (k0 + TK < K) {
      int nb = buf ^ 1;
      async_b128(ga + k0 + TK,     &Al[nb][r * TK]);
      async_b128(ga + k0 + TK + 8, &Al[nb][r * TK + 8]);
      async_b128(ga + k0 + TK + 16, &Al[nb][r * TK + 16]);
      async_b128(ga + k0 + TK + 24, &Al[nb][r * TK + 24]);
      async_b128(gb + k0 + TK,     &Bl[nb][r * TK]);
      async_b128(gb + k0 + TK + 8, &Bl[nb][r * TK + 8]);
      async_b128(gb + k0 + TK + 16, &Bl[nb][r * TK + 16]);
      async_b128(gb + k0 + TK + 24, &Bl[nb][r * TK + 24]);
    }
    wave_mma(Al[buf], Bl[buf], wr, wc, l16, hi, acc);
    buf ^= 1;
  }
#else
  for (int k0 = 0; k0 < K; k0 += TK) {
    *(uint4*)&Al[0][r * TK]     = *(const uint4*)(ga + k0);
    *(uint4*)&Al[0][r * TK + 8] = *(const uint4*)(ga + k0 + 8);
    *(uint4*)&Al[0][r * TK + 16] = *(const uint4*)(ga + k0 + 16);
    *(uint4*)&Al[0][r * TK + 24] = *(const uint4*)(ga + k0 + 24);
    *(uint4*)&Bl[0][r * TK]     = *(const uint4*)(gb + k0);
    *(uint4*)&Bl[0][r * TK + 8] = *(const uint4*)(gb + k0 + 8);
    *(uint4*)&Bl[0][r * TK + 16] = *(const uint4*)(gb + k0 + 16);
    *(uint4*)&Bl[0][r * TK + 24] = *(const uint4*)(gb + k0 + 24);
    __syncthreads();
    wave_mma(Al[0], Bl[0], wr, wc, l16, hi, acc);
    __syncthreads();
  }
#endif
}

// ---------------------------------------------------------------- zero
__global__ void k_zero(int* counts) {
  if (threadIdx.x < 16) counts[threadIdx.x] = 0;  // counts[8] + 8 pad ints; cursor is reset in k_scan
}

// ---------------------------------------------------------------- weight convert+transpose
// src: [E][K][N] f32 (blockIdx.z = expert) -> dst: [E][N][K] bf16
__global__ void __launch_bounds__(256) k_wconv(const float* __restrict__ src,
                                               unsigned short* __restrict__ dst,
                                               int K, int N) {
  __shared__ unsigned short t[32][33];
  size_t eo = (size_t)blockIdx.z * K * N;
  int k0 = blockIdx.y * 32, n0 = blockIdx.x * 32;
  int tx = threadIdx.x & 31, ty = threadIdx.x >> 5;   // 32 x 8
#pragma unroll
  for (int i = 0; i < 32; i += 8)
    t[ty + i][tx] = f2bf(src[eo + (size_t)(k0 + ty + i) * N + n0 + tx]);
  __syncthreads();
#pragma unroll
  for (int i = 0; i < 32; i += 8)
    dst[eo + (size_t)(n0 + ty + i) * K + k0 + tx] = t[tx][ty + i];
}

// ---------------------------------------------------------------- router
__global__ void __launch_bounds__(256) k_router(
    const float* __restrict__ x, const float* __restrict__ gw,
    float* __restrict__ logits_out, unsigned short* __restrict__ xb,
    int* __restrict__ topi, float* __restrict__ topw,
    int* __restrict__ counts, int B) {
  __shared__ float gws[Ec * Dc];                       // 32 KB
  int tid = threadIdx.x;
  for (int i = tid * 4; i < Ec * Dc; i += 256 * 4)
    *(float4*)&gws[i] = *(const float4*)&gw[i];
  __syncthreads();

  int b = blockIdx.x * 256 + tid;
  if (b >= B) return;
  const float* xr = x + (size_t)b * Dc;

  float acc[Ec];
#pragma unroll
  for (int e = 0; e < Ec; ++e) acc[e] = 0.f;

  for (int d = 0; d < Dc; d += 4) {
    float4 xv = *(const float4*)(xr + d);
    uint2 pk;
    pk.x = (unsigned)f2bf(xv.x) | ((unsigned)f2bf(xv.y) << 16);
    pk.y = (unsigned)f2bf(xv.z) | ((unsigned)f2bf(xv.w) << 16);
    *(uint2*)&xb[(size_t)b * Dc + d] = pk;
#pragma unroll
    for (int e = 0; e < Ec; ++e) {
      float4 g = *(const float4*)&gws[e * Dc + d];
      acc[e] += xv.x * g.x + xv.y * g.y + xv.z * g.z + xv.w * g.w;
    }
  }

  float m = acc[0];
#pragma unroll
  for (int e = 1; e < Ec; ++e) m = fmaxf(m, acc[e]);
  float p[Ec];
#pragma unroll
  for (int e = 0; e < Ec; ++e) {
    p[e] = __expf(acc[e] - m);
    logits_out[(size_t)b * Ec + e] = acc[e];
  }
  int i0 = 0; float p0 = p[0];
#pragma unroll
  for (int e = 1; e < Ec; ++e) if (p[e] > p0) { p0 = p[e]; i0 = e; }
  int i1 = (i0 == 0) ? 1 : 0; float p1 = p[i1];
#pragma unroll
  for (int e = 0; e < Ec; ++e)
    if (e != i0 && p[e] > p1) { p1 = p[e]; i1 = e; }

  float inv = 1.f / (p0 + p1);
  topi[2 * b]     = i0;  topi[2 * b + 1] = i1;
  topw[2 * b]     = p0 * inv;
  topw[2 * b + 1] = p1 * inv;
  atomicAdd(&counts[i0], 1);
  atomicAdd(&counts[i1], 1);
}

// ---------------------------------------------------------------- scan
__global__ void k_scan(const int* __restrict__ counts, int* __restrict__ offs,
                       int* __restrict__ cursor) {
  if (threadIdx.x == 0) {
    int s = 0;
    for (int e = 0; e < Ec; ++e) { offs[e] = s; s += counts[e]; cursor[e] = 0; }
    offs[Ec] = s;
  }
}

// ---------------------------------------------------------------- placement
__global__ void k_place(const int* __restrict__ topi, const int* __restrict__ offs,
                        int* __restrict__ cursor, int* __restrict__ tok_idx,
                        int* __restrict__ assign_row, int B) {
  int b = blockIdx.x * 256 + threadIdx.x;
  if (b >= B) return;
#pragma unroll
  for (int k = 0; k < 2; ++k) {
    int e = topi[2 * b + k];
    int pos = offs[e] + atomicAdd(&cursor[e], 1);
    tok_idx[pos] = b;
    assign_row[2 * b + k] = pos;
  }
}

// ---------------------------------------------------------------- GEMM1: h = relu(bn(x@w1 + b1))
__global__ void __launch_bounds__(128) k_mlp1(
    const unsigned short* __restrict__ xb, const int* __restrict__ tok_idx,
    const int* __restrict__ offs, const unsigned short* __restrict__ w1t,
    const float* __restrict__ b1, const float* __restrict__ gamma,
    const float* __restrict__ beta, const float* __restrict__ mean,
    const float* __restrict__ var, unsigned short* __restrict__ h, int mtiles) {
  int e = blockIdx.y / mtiles;
  int mt = blockIdx.y % mtiles;
  int base = offs[e], end = offs[e + 1];
  if (mt * TM >= end - base) return;

  __shared__ unsigned short Al[NBUF][TM * TK];   // [m][k]
  __shared__ unsigned short Bl[NBUF][TN * TK];   // [n][k]

  int tid = threadIdx.x;
  int lane = tid & 31, wave = tid >> 5;
  int wr = wave >> 1, wc = wave & 1;
  int l16 = lane & 15, hi = lane >> 4;
  int n0 = blockIdx.x * TN;

  int r = tid;
  int gp_r = base + mt * TM + r;
  int clamped = gp_r < end ? gp_r : end - 1;
  const unsigned short* ga = xb + (size_t)tok_idx[clamped] * Dc;
  const unsigned short* gb = w1t + ((size_t)e * Hc + n0 + r) * Dc;   // [E][H][D] bf16

  v8f acc[16] = {};
  gemm_mainloop<Dc>(ga, gb, Al, Bl, r, wr, wc, l16, hi, acc);

#pragma unroll
  for (int ni = 0; ni < 4; ++ni) {
    int ng = n0 + wc * 64 + ni * 16 + l16;
    size_t pe = (size_t)e * Hc + ng;
    float iv  = gamma[pe] * rsqrtf(var[pe] + BN_EPS);
    float sh  = beta[pe] - mean[pe] * iv;
    float bia = b1[pe];
#pragma unroll
    for (int mi = 0; mi < 4; ++mi) {
      int mb = wr * 64 + mi * 16 + (hi ? 8 : 0);
      v8f a = acc[mi * 4 + ni];
#pragma unroll
      for (int v = 0; v < 8; ++v) {
        int gp = base + mt * TM + mb + v;
        if (gp < end) {
          float hv = fmaf(a[v] + bia, iv, sh);
          hv = hv > 0.f ? hv : 0.f;
          h[(size_t)gp * Hc + ng] = f2bf(hv);
        }
      }
    }
  }
}

// ---------------------------------------------------------------- GEMM2: y = h@w2 + b2
__global__ void __launch_bounds__(128) k_mlp2(
    const unsigned short* __restrict__ h, const int* __restrict__ offs,
    const unsigned short* __restrict__ w2t, const float* __restrict__ b2,
    float* __restrict__ y, int mtiles) {
  int e = blockIdx.y / mtiles;
  int mt = blockIdx.y % mtiles;
  int base = offs[e], end = offs[e + 1];
  if (mt * TM >= end - base) return;

  __shared__ unsigned short Al[NBUF][TM * TK];
  __shared__ unsigned short Bl[NBUF][TN * TK];

  int tid = threadIdx.x;
  int lane = tid & 31, wave = tid >> 5;
  int wr = wave >> 1, wc = wave & 1;
  int l16 = lane & 15, hi = lane >> 4;
  int n0 = blockIdx.x * TN;

  int r = tid;
  int gp_r = base + mt * TM + r;
  int clamped = gp_r < end ? gp_r : end - 1;
  const unsigned short* ga = h + (size_t)clamped * Hc;
  const unsigned short* gb = w2t + ((size_t)e * Dc + n0 + r) * Hc;   // [E][D][H] bf16

  v8f acc[16] = {};
  gemm_mainloop<Hc>(ga, gb, Al, Bl, r, wr, wc, l16, hi, acc);

#pragma unroll
  for (int ni = 0; ni < 4; ++ni) {
    int ng = n0 + wc * 64 + ni * 16 + l16;
    float bia = b2[(size_t)e * Dc + ng];
#pragma unroll
    for (int mi = 0; mi < 4; ++mi) {
      int mb = wr * 64 + mi * 16 + (hi ? 8 : 0);
      v8f a = acc[mi * 4 + ni];
#pragma unroll
      for (int v = 0; v < 8; ++v) {
        int gp = base + mt * TM + mb + v;
        if (gp < end) y[(size_t)gp * Dc + ng] = a[v] + bia;
      }
    }
  }
}

// ---------------------------------------------------------------- combine
__global__ void __launch_bounds__(256) k_combine(
    const float* __restrict__ x, const float* __restrict__ y,
    const int* __restrict__ assign_row, const float* __restrict__ topw,
    float* __restrict__ out, int B) {
  int idx = blockIdx.x * 256 + threadIdx.x;       // float4 index
  int total = B * (Dc / 4);
  if (idx >= total) return;
  int b  = idx / (Dc / 4);
  int dq = idx % (Dc / 4);
  int r0 = assign_row[2 * b], r1 = assign_row[2 * b + 1];
  float w0 = topw[2 * b], w1v = topw[2 * b + 1];
  float4 xv = ((const float4*)x)[idx];
  float4 y0 = ((const float4*)(y + (size_t)r0 * Dc))[dq];
  float4 y1 = ((const float4*)(y + (size_t)r1 * Dc))[dq];
  float4 o;
  o.x = fmaf(w0, y0.x, fmaf(w1v, y1.x, xv.x));
  o.y = fmaf(w0, y0.y, fmaf(w1v, y1.y, xv.y));
  o.z = fmaf(w0, y0.z, fmaf(w1v, y1.z, xv.z));
  o.w = fmaf(w0, y0.w, fmaf(w1v, y1.w, xv.w));
  ((float4*)out)[idx] = o;
}

// ---------------------------------------------------------------- launch
extern "C" void kernel_launch(void* const* d_in, const int* in_sizes, int n_in,
                              void* d_out, int out_size, void* d_ws, size_t ws_size,
                              hipStream_t stream) {
  const float* x      = (const float*)d_in[0];
  const float* gate_w = (const float*)d_in[1];
  const float* w1     = (const float*)d_in[2];
  const float* b1     = (const float*)d_in[3];
  const float* gamma  = (const float*)d_in[4];
  const float* beta   = (const float*)d_in[5];
  const float* mean   = (const float*)d_in[6];
  const float* var    = (const float*)d_in[7];
  const float* w2     = (const float*)d_in[8];
  const float* b2     = (const float*)d_in[9];

  int B = in_sizes[0] / Dc;
  float* out    = (float*)d_out;
  float* logits = (float*)d_out + (size_t)B * Dc;

  char* w = (char*)d_ws;
  size_t off = 0;
  unsigned short* xb = (unsigned short*)(w + off); off += (size_t)B * Dc * 2;
  int*   counts = (int*)(w + off);                 // [8]
  int*   cursor = (int*)(w + off + 64);            // [8]
  int*   offs   = (int*)(w + off + 128);           // [9]
  off += 256;
  int*   topi   = (int*)(w + off);   off += (size_t)2 * B * 4;
  float* topw   = (float*)(w + off); off += (size_t)2 * B * 4;
  int*   tokidx = (int*)(w + off);   off += (size_t)2 * B * 4;
  int*   assign = (int*)(w + off);   off += (size_t)2 * B * 4;
  unsigned short* w1t = (unsigned short*)(w + off); off += (size_t)Ec * Dc * Hc * 2;
  unsigned short* w2t = (unsigned short*)(w + off); off += (size_t)Ec * Hc * Dc * 2;
  unsigned short* hbuf = (unsigned short*)(w + off); off += (size_t)2 * B * Hc * 2;
  float* ybuf = (float*)(w + off);   off += (size_t)2 * B * Dc * 4;

  int mtiles = (B + TM - 1) / TM;

  k_zero<<<1, 32, 0, stream>>>(counts);
  // weight convert+transpose: w1 [E][D][H] -> w1t [E][H][D]; w2 [E][H][D] -> w2t [E][D][H]
  k_wconv<<<dim3(Hc / 32, Dc / 32, Ec), 256, 0, stream>>>(w1, w1t, Dc, Hc);
  k_wconv<<<dim3(Dc / 32, Hc / 32, Ec), 256, 0, stream>>>(w2, w2t, Hc, Dc);
  k_router<<<(B + 255) / 256, 256, 0, stream>>>(x, gate_w, logits, xb, topi, topw,
                                                counts, B);
  k_scan<<<1, 1, 0, stream>>>(counts, offs, cursor);
  k_place<<<(B + 255) / 256, 256, 0, stream>>>(topi, offs, cursor, tokidx, assign, B);
  k_mlp1<<<dim3(Hc / TN, Ec * mtiles), 128, 0, stream>>>(
      xb, tokidx, offs, w1t, b1, gamma, beta, mean, var, hbuf, mtiles);
  k_mlp2<<<dim3(Dc / TN, Ec * mtiles), 128, 0, stream>>>(
      hbuf, offs, w2t, b2, ybuf, mtiles);
  k_combine<<<(B * (Dc / 4) + 255) / 256, 256, 0, stream>>>(x, ybuf, assign, topw,
                                                            out, B);
}
